// TargetEnhanceModule_47425028882817
// MI455X (gfx1250) — matrix operand order repaired and checked by reading, not compiled, not run on hardware
//
#include <hip/hip_runtime.h>

// ---------------------------------------------------------------------------
// Problem constants (match reference)
// ---------------------------------------------------------------------------
#define BATCH 8
#define CDIM  256
#define HDIM  64
#define WDIM  64
#define HW    4096          // HDIM*WDIM
#define EPSI  1e-5f

typedef __attribute__((ext_vector_type(16))) __bf16 v16bf;
typedef __attribute__((ext_vector_type(8)))  float  v8f;

union FragU { v16bf bf; uint4 u[2]; };

__device__ __forceinline__ v8f wmma_bf16(v16bf a, v16bf b, v8f c) {
    // (neg_a, A, neg_b, B, c_mod, C, reuse_a, reuse_b)
    return __builtin_amdgcn_wmma_f32_16x16x32_bf16(false, a, false, b,
                                                   (short)0, c, false, false);
}

// ---------------------------------------------------------------------------
// K0: convert the three 256x256 f32 weight matrices to bf16
// ---------------------------------------------------------------------------
__global__ __launch_bounds__(256) void prep_kernel(
    const float* __restrict__ wq, const float* __restrict__ wk,
    const float* __restrict__ wv, __bf16* __restrict__ wbf)
{
    int idx  = blockIdx.x * 256 + threadIdx.x;     // 0 .. 3*65536-1
    int proj = idx >> 16;
    int off  = idx & 65535;
    const float* w = (proj == 0) ? wq : (proj == 1) ? wk : wv;
    wbf[idx] = (__bf16)w[off];
}

// ---------------------------------------------------------------------------
// K1: projection GEMM.  Block = 256 thr (8 waves).  Each block computes all
// 256 output channels for a 16-position tile of one (proj, batch).
// A = W (M=o, K=c, c-contiguous), B = X tile staged in LDS as [p][c] bf16.
// q/k written [b][p][o], v written [b][o][p].
// ---------------------------------------------------------------------------
__global__ __launch_bounds__(256) void proj_kernel(
    const float* __restrict__ x,            // [B][C][HW]
    const float* __restrict__ sal,          // [B][HW]
    const __bf16* __restrict__ wbf,         // [3][256][256]
    const float* __restrict__ bq, const float* __restrict__ bk,
    const float* __restrict__ bv,
    __bf16* __restrict__ yq, __bf16* __restrict__ yk, __bf16* __restrict__ yv)
{
    int bid   = blockIdx.x;                 // 3*B*(HW/16)
    int ptile = bid & 255;                  // HW/16 == 256
    int tmp   = bid >> 8;
    int b     = tmp & 7;
    int proj  = tmp >> 3;
    int p0    = ptile * 16;

    __shared__ __bf16 Xs[16][CDIM];         // 8 KB

    int t = threadIdx.x;
    const float* xb = x + (size_t)b * CDIM * HW;
    bool masked = (proj == 0);
    #pragma unroll
    for (int i = t; i < 16 * CDIM; i += 256) {
        int c = i >> 4, p = i & 15;
        float v = xb[(size_t)c * HW + p0 + p];
        if (masked) v *= sal[(size_t)b * HW + p0 + p];
        Xs[p][c] = (__bf16)v;
    }
    __syncthreads();

    int wave = t >> 5, lane = t & 31, lh = lane >> 4, ll = lane & 15;
    int o_base = wave * 32;                 // wave covers 32 output channels

    const __bf16* W = wbf + (size_t)proj * CDIM * CDIM;
    v8f acc0 = {}, acc1 = {};
    #pragma unroll
    for (int c0 = 0; c0 < CDIM; c0 += 32) {
        FragU bf;
        bf.u[0] = *(const uint4*)&Xs[ll][c0 + lh * 8];
        bf.u[1] = *(const uint4*)&Xs[ll][c0 + 16 + lh * 8];
        const __bf16* wr0 = W + (size_t)(o_base + ll) * CDIM + c0;
        const __bf16* wr1 = W + (size_t)(o_base + 16 + ll) * CDIM + c0;
        FragU a0, a1;
        a0.u[0] = *(const uint4*)(wr0 + lh * 8);
        a0.u[1] = *(const uint4*)(wr0 + 16 + lh * 8);
        a1.u[0] = *(const uint4*)(wr1 + lh * 8);
        a1.u[1] = *(const uint4*)(wr1 + 16 + lh * 8);
        acc0 = wmma_bf16(a0.bf, bf.bf, acc0);
        acc1 = wmma_bf16(a1.bf, bf.bf, acc1);
    }

    const float* bias = (proj == 0) ? bq : (proj == 1) ? bk : bv;
    __bf16* Y = (proj == 0) ? yq : (proj == 1) ? yk : yv;
    int p = p0 + ll;
    #pragma unroll
    for (int i = 0; i < 8; ++i) {
        int o = o_base + i + 8 * lh;
        float v0 = acc0[i] + bias[o];
        size_t i0 = (proj < 2) ? (((size_t)b * HW + p) * CDIM + o)
                               : (((size_t)b * CDIM + o) * HW + p);
        Y[i0] = (__bf16)v0;
        int o1 = o + 16;
        float v1 = acc1[i] + bias[o1];
        size_t i1 = (proj < 2) ? (((size_t)b * HW + p) * CDIM + o1)
                               : (((size_t)b * CDIM + o1) * HW + p);
        Y[i1] = (__bf16)v1;
    }
}

// ---------------------------------------------------------------------------
// K2: per-channel BN stats over (B,H,W); fold into scale a and shift c.
// grid = 3*256 blocks, 256 threads each.
// ---------------------------------------------------------------------------
__global__ __launch_bounds__(256) void stats_kernel(
    const __bf16* __restrict__ yq, const __bf16* __restrict__ yk,
    const __bf16* __restrict__ yv,
    const float* gq, const float* btq, const float* gk, const float* btk,
    const float* gv, const float* btv,
    float* __restrict__ scale_a, float* __restrict__ shift_c)
{
    int ch   = blockIdx.x & 255;
    int proj = blockIdx.x >> 8;
    const __bf16* Y = (proj == 0) ? yq : (proj == 1) ? yk : yv;
    int t = threadIdx.x;
    const int total = BATCH * HW;
    float s = 0.f, s2 = 0.f;
    if (proj < 2) {                         // layout [b][p][ch]
        for (int i = t; i < total; i += 256) {
            float v = (float)Y[(size_t)i * CDIM + ch];
            s += v; s2 += v * v;
        }
    } else {                                // layout [b][ch][p]
        for (int b = 0; b < BATCH; ++b) {
            const __bf16* base = Y + ((size_t)b * CDIM + ch) * HW;
            for (int i = t; i < HW; i += 256) {
                float v = (float)base[i];
                s += v; s2 += v * v;
            }
        }
    }
    __shared__ float ss[256], sq[256];
    ss[t] = s; sq[t] = s2;
    __syncthreads();
    for (int st = 128; st > 0; st >>= 1) {
        if (t < st) { ss[t] += ss[t + st]; sq[t] += sq[t + st]; }
        __syncthreads();
    }
    if (t == 0) {
        float mean = ss[0] / (float)total;
        float var  = sq[0] / (float)total - mean * mean;
        const float* g  = (proj == 0) ? gq  : (proj == 1) ? gk  : gv;
        const float* bt = (proj == 0) ? btq : (proj == 1) ? btk : btv;
        float a = g[ch] * rsqrtf(var + EPSI);
        scale_a[blockIdx.x] = a;
        shift_c[blockIdx.x] = bt[ch] - mean * a;
    }
}

// ---------------------------------------------------------------------------
// K3: apply BN in place.  grid = dim3(per/256, 3)
// ---------------------------------------------------------------------------
__global__ __launch_bounds__(256) void apply_kernel(
    __bf16* __restrict__ yq, __bf16* __restrict__ yk, __bf16* __restrict__ yv,
    const float* __restrict__ sa, const float* __restrict__ sc)
{
    int proj = blockIdx.y;
    size_t idx = (size_t)blockIdx.x * 256 + threadIdx.x;
    __bf16* Y = (proj == 0) ? yq : (proj == 1) ? yk : yv;
    int ch = (proj < 2) ? (int)(idx & 255) : (int)((idx >> 12) & 255);
    float a = sa[proj * 256 + ch], c = sc[proj * 256 + ch];
    Y[idx] = (__bf16)(a * (float)Y[idx] + c);
}

// ---------------------------------------------------------------------------
// K4: softmax row statistics.  Block = 8 waves, owns 16 rows (n) of one batch,
// streams all m via WMMA score tiles.  Pass 1 row max, pass 2 sum(exp).
// ---------------------------------------------------------------------------
__global__ __launch_bounds__(256) void rowstats_kernel(
    const __bf16* __restrict__ Qb,          // [B][n][c]
    const __bf16* __restrict__ Kb,          // [B][m][c]
    float* __restrict__ rowmax, float* __restrict__ rowinv)
{
    int ntile = blockIdx.x & 255;           // HW/16 == 256
    int b     = blockIdx.x >> 8;
    int n0    = ntile * 16;
    int t = threadIdx.x, wave = t >> 5, lane = t & 31, lh = lane >> 4, ll = lane & 15;

    const __bf16* Q = Qb + (size_t)b * HW * CDIM;
    const __bf16* K = Kb + (size_t)b * HW * CDIM;

    // Preload the 8 Q A-frags (whole K=256) — reused by both passes.
    FragU afr[8];
    {
        const __bf16* qr = Q + (size_t)(n0 + ll) * CDIM;
        #pragma unroll
        for (int ks = 0; ks < 8; ++ks) {
            int c0 = ks * 32;
            afr[ks].u[0] = *(const uint4*)(qr + c0 + lh * 8);
            afr[ks].u[1] = *(const uint4*)(qr + c0 + 16 + lh * 8);
        }
    }

    __shared__ float sred[8][16];
    __shared__ float rowm[16];

    // ---- pass 1: row max ----
    float rmax[8];
    #pragma unroll
    for (int i = 0; i < 8; ++i) rmax[i] = -3.0e38f;
    for (int m0 = wave * 16; m0 < HW; m0 += 128) {
        const __bf16* kr = K + (size_t)(m0 + ll) * CDIM;
        v8f acc = {};
        #pragma unroll
        for (int ks = 0; ks < 8; ++ks) {
            int c0 = ks * 32;
            FragU bf;
            bf.u[0] = *(const uint4*)(kr + c0 + lh * 8);
            bf.u[1] = *(const uint4*)(kr + c0 + 16 + lh * 8);
            acc = wmma_bf16(afr[ks].bf, bf.bf, acc);
        }
        #pragma unroll
        for (int i = 0; i < 8; ++i) rmax[i] = fmaxf(rmax[i], acc[i]);
    }
    #pragma unroll
    for (int m = 1; m < 16; m <<= 1)
        #pragma unroll
        for (int i = 0; i < 8; ++i)
            rmax[i] = fmaxf(rmax[i], __shfl_xor(rmax[i], m, 32));
    if (ll == 0)
        #pragma unroll
        for (int i = 0; i < 8; ++i) sred[wave][i + 8 * lh] = rmax[i];
    __syncthreads();
    if (t < 16) {
        float v = sred[0][t];
        for (int w2 = 1; w2 < 8; ++w2) v = fmaxf(v, sred[w2][t]);
        rowm[t] = v;
    }
    __syncthreads();
    float rm[8];
    #pragma unroll
    for (int i = 0; i < 8; ++i) rm[i] = rowm[i + 8 * lh];
    __syncthreads();

    // ---- pass 2: sum(exp) ----
    float rsum[8];
    #pragma unroll
    for (int i = 0; i < 8; ++i) rsum[i] = 0.f;
    for (int m0 = wave * 16; m0 < HW; m0 += 128) {
        const __bf16* kr = K + (size_t)(m0 + ll) * CDIM;
        v8f acc = {};
        #pragma unroll
        for (int ks = 0; ks < 8; ++ks) {
            int c0 = ks * 32;
            FragU bf;
            bf.u[0] = *(const uint4*)(kr + c0 + lh * 8);
            bf.u[1] = *(const uint4*)(kr + c0 + 16 + lh * 8);
            acc = wmma_bf16(afr[ks].bf, bf.bf, acc);
        }
        #pragma unroll
        for (int i = 0; i < 8; ++i) rsum[i] += __expf(acc[i] - rm[i]);
    }
    #pragma unroll
    for (int m = 1; m < 16; m <<= 1)
        #pragma unroll
        for (int i = 0; i < 8; ++i) rsum[i] += __shfl_xor(rsum[i], m, 32);
    if (ll == 0)
        #pragma unroll
        for (int i = 0; i < 8; ++i) sred[wave][i + 8 * lh] = rsum[i];
    __syncthreads();
    if (t < 16) {
        float v = 0.f;
        for (int w2 = 0; w2 < 8; ++w2) v += sred[w2][t];
        rowmax[(size_t)b * HW + n0 + t] = rowm[t];
        rowinv[(size_t)b * HW + n0 + t] = 1.0f / v;
    }
}

// ---------------------------------------------------------------------------
// K5: fused attention output.  Block = 8 waves covering 64 m-positions x all
// 256 channels of one batch.  wave = (m-tile 0..3, c-half 0..1).  For each
// 32-wide n-chunk: WMMA scores -> exp/scale -> LDS transpose -> WMMA V x P.
// ---------------------------------------------------------------------------
__global__ __launch_bounds__(256) void attnout_kernel(
    const __bf16* __restrict__ Qb,          // [B][n][c]
    const __bf16* __restrict__ Kb,          // [B][m][c]
    const __bf16* __restrict__ Vb,          // [B][c][n]
    const float* __restrict__ rowmax, const float* __restrict__ rowinv,
    float* __restrict__ out)                // [B][C][HW]
{
    int mblk = blockIdx.x & 63;             // HW/64 == 64
    int b    = blockIdx.x >> 6;
    int t = threadIdx.x, wave = t >> 5, lane = t & 31, lh = lane >> 4, ll = lane & 15;
    int mt    = wave >> 1;                  // 0..3
    int chalf = wave & 1;                   // 0..1
    int m0    = mblk * 64 + mt * 16;
    int cbase = chalf * 128;

    const __bf16* Q = Qb + (size_t)b * HW * CDIM;
    const __bf16* K = Kb + (size_t)b * HW * CDIM;
    const __bf16* V = Vb + (size_t)b * CDIM * HW;
    const float* rmaxb = rowmax + (size_t)b * HW;
    const float* rinvb = rowinv + (size_t)b * HW;

    __shared__ __bf16 Ps[8][16][32];        // per-wave [m][n] P tile, 8 KB

    // Preload K B-frags for this wave's m-tile (constant over n-loop).
    FragU kfr[8];
    {
        const __bf16* kr = K + (size_t)(m0 + ll) * CDIM;
        #pragma unroll
        for (int ks = 0; ks < 8; ++ks) {
            int c0 = ks * 32;
            kfr[ks].u[0] = *(const uint4*)(kr + c0 + lh * 8);
            kfr[ks].u[1] = *(const uint4*)(kr + c0 + 16 + lh * 8);
        }
    }

    v8f acc[8];
    #pragma unroll
    for (int ct = 0; ct < 8; ++ct) acc[ct] = (v8f){};

    for (int n0 = 0; n0 < HW; n0 += 32) {
        // scores s[n, m] for two 16-row n-tiles
        v8f s0 = {}, s1 = {};
        const __bf16* qr0 = Q + (size_t)(n0 + ll) * CDIM;
        const __bf16* qr1 = Q + (size_t)(n0 + 16 + ll) * CDIM;
        #pragma unroll
        for (int ks = 0; ks < 8; ++ks) {
            int c0 = ks * 32;
            FragU a0, a1;
            a0.u[0] = *(const uint4*)(qr0 + c0 + lh * 8);
            a0.u[1] = *(const uint4*)(qr0 + c0 + 16 + lh * 8);
            a1.u[0] = *(const uint4*)(qr1 + c0 + lh * 8);
            a1.u[1] = *(const uint4*)(qr1 + c0 + 16 + lh * 8);
            s0 = wmma_bf16(a0.bf, kfr[ks].bf, s0);
            s1 = wmma_bf16(a1.bf, kfr[ks].bf, s1);
        }
        // exp / scale, transpose into LDS as [m][n]
        #pragma unroll
        for (int i = 0; i < 8; ++i) {
            int nr = i + 8 * lh;
            int n  = n0 + nr;
            float p0v = __expf(s0[i] - rmaxb[n]) * rinvb[n];
            float p1v = __expf(s1[i] - rmaxb[n + 16]) * rinvb[n + 16];
            Ps[wave][ll][nr]      = (__bf16)p0v;
            Ps[wave][ll][nr + 16] = (__bf16)p1v;
        }
        // P B-frag (K = n, 32-wide), per-wave private LDS (in-wave ordering)
        FragU pf;
        pf.u[0] = *(const uint4*)&Ps[wave][ll][lh * 8];
        pf.u[1] = *(const uint4*)&Ps[wave][ll][16 + lh * 8];
        // out += V[c, n-chunk] x P
        #pragma unroll
        for (int ct = 0; ct < 8; ++ct) {
            const __bf16* vr =
                V + (size_t)(cbase + ct * 16 + ll) * HW + n0;
            FragU av;
            av.u[0] = *(const uint4*)(vr + lh * 8);
            av.u[1] = *(const uint4*)(vr + 16 + lh * 8);
            acc[ct] = wmma_bf16(av.bf, pf.bf, acc[ct]);
        }
    }

    // store: out[b][c][m], f32
    #pragma unroll
    for (int ct = 0; ct < 8; ++ct) {
        #pragma unroll
        for (int i = 0; i < 8; ++i) {
            int c = cbase + ct * 16 + i + 8 * lh;
            int m = m0 + ll;
            out[((size_t)b * CDIM + c) * HW + m] = acc[ct][i];
        }
    }
}

// ---------------------------------------------------------------------------
// Host launcher
// ---------------------------------------------------------------------------
extern "C" void kernel_launch(void* const* d_in, const int* in_sizes, int n_in,
                              void* d_out, int out_size, void* d_ws, size_t ws_size,
                              hipStream_t stream) {
    const float* x    = (const float*)d_in[0];
    const float* sal  = (const float*)d_in[1];
    const float* wq   = (const float*)d_in[2];
    const float* bq   = (const float*)d_in[3];
    const float* gq   = (const float*)d_in[4];
    const float* btq  = (const float*)d_in[5];
    const float* wk   = (const float*)d_in[6];
    const float* bk   = (const float*)d_in[7];
    const float* gk   = (const float*)d_in[8];
    const float* btk  = (const float*)d_in[9];
    const float* wv   = (const float*)d_in[10];
    const float* bv   = (const float*)d_in[11];
    const float* gv   = (const float*)d_in[12];
    const float* btv  = (const float*)d_in[13];
    float* out        = (float*)d_out;

    // workspace carve-up (≈ 49 MiB total)
    const size_t YBYTES = (size_t)BATCH * HW * CDIM * sizeof(__bf16); // 16 MiB
    char* ws = (char*)d_ws;
    __bf16* yq  = (__bf16*)ws;                 ws += YBYTES;
    __bf16* yk  = (__bf16*)ws;                 ws += YBYTES;
    __bf16* yv  = (__bf16*)ws;                 ws += YBYTES;
    __bf16* wbf = (__bf16*)ws;                 ws += (size_t)3 * CDIM * CDIM * sizeof(__bf16);
    float*  sa  = (float*)ws;                  ws += 3 * 256 * sizeof(float);
    float*  sc  = (float*)ws;                  ws += 3 * 256 * sizeof(float);
    float*  rmx = (float*)ws;                  ws += (size_t)BATCH * HW * sizeof(float);
    float*  rin = (float*)ws;

    // K0: weights -> bf16
    prep_kernel<<<dim3(768), dim3(256), 0, stream>>>(wq, wk, wv, wbf);

    // K1: projections (WMMA)
    proj_kernel<<<dim3(3 * BATCH * (HW / 16)), dim3(256), 0, stream>>>(
        x, sal, wbf, bq, bk, bv, yq, yk, yv);

    // K2: BN stats -> scale/shift
    stats_kernel<<<dim3(3 * 256), dim3(256), 0, stream>>>(
        yq, yk, yv, gq, btq, gk, btk, gv, btv, sa, sc);

    // K3: apply BN in place
    apply_kernel<<<dim3((BATCH * HW * CDIM) / 256, 3), dim3(256), 0, stream>>>(
        yq, yk, yv, sa, sc);

    // K4: softmax row statistics (WMMA score streaming)
    rowstats_kernel<<<dim3(BATCH * (HW / 16)), dim3(256), 0, stream>>>(
        yq, yk, rmx, rin);

    // K5: fused attention output (WMMA scores + WMMA V x P)
    attnout_kernel<<<dim3(BATCH * (HW / 64)), dim3(256), 0, stream>>>(
        yq, yk, yv, rmx, rin, out);
}
